// TriangleSelfAttention_16999480558056
// MI455X (gfx1250) — matrix operand order repaired and checked by reading, not compiled, not run on hardware
//
#include <hip/hip_runtime.h>
#include <hip/hip_bf16.h>

// ---------------------------------------------------------------------------
// Triangle self-attention for MI455X (gfx1250, wave32, WMMA 16x16x32 f16).
// B=1, DZ=128, L=256, H=4, DC=32, DHC=128, tokens NT = L*L = 65536.
// ---------------------------------------------------------------------------

typedef __attribute__((ext_vector_type(16))) _Float16 v16h;
typedef __attribute__((ext_vector_type(8)))  _Float16 v8h;
typedef __attribute__((ext_vector_type(8)))  float    v8f;
typedef int i4v __attribute__((vector_size(16)));

#define LSEQ 256
#define DZC  128
#define NT   65536   // LSEQ*LSEQ tokens
#define NHEAD 4
#define DCH  32

// Async global->LDS copy (CDNA5 ASYNCcnt path), guarded for toolchain support.
#if defined(__has_builtin)
#  if __has_builtin(__builtin_amdgcn_global_load_async_to_lds_b128)
#    define USE_ASYNC_LDS 1
#  endif
#endif
#ifndef USE_ASYNC_LDS
#  define USE_ASYNC_LDS 0
#endif

// A-matrix fragment (16x32 f16, M x K): lane<16 -> M=lane, K = base + e (+8 for e>=8)
// memory row-major [M][K] contiguous -> lo = p[0..7], hi = p[16..23]
static __device__ inline v16h ld_a16(const _Float16* p) {
  v8h lo = *(const v8h*)p;
  v8h hi = *(const v8h*)(p + 16);
  v16h r;
#pragma unroll
  for (int e = 0; e < 8; ++e) { r[e] = lo[e]; r[e + 8] = hi[e]; }
  return r;
}

// 16 contiguous halves (B fragments pre-swizzled)
static __device__ inline v16h ld16(const _Float16* p) {
  v8h lo = *(const v8h*)p;
  v8h hi = *(const v8h*)(p + 8);
  v16h r;
#pragma unroll
  for (int e = 0; e < 8; ++e) { r[e] = lo[e]; r[e + 8] = hi[e]; }
  return r;
}

static __device__ inline v8f wmma_f16(v16h a, v16h b, v8f c) {
  return __builtin_amdgcn_wmma_f32_16x16x32_f16(false, a, false, b, (short)0, c,
                                                false, false);
}

// ---------------------------------------------------------------------------
// K0: swizzle 5 weights (128x128 f32, row-major [K][N]) into WMMA-B fragment
// order: frag[((nb*4+kc)*32 + lane)*16 + e] = w[(kc*32 + (e>>3)*16 + (lane>>4)*8 + (e&7))*128
//                                              + nb*16 + (lane&15)]
// ---------------------------------------------------------------------------
__global__ void convert_weights_kernel(const float* __restrict__ wq,
                                       const float* __restrict__ wk,
                                       const float* __restrict__ wv,
                                       const float* __restrict__ wg,
                                       const float* __restrict__ w2,
                                       _Float16* __restrict__ frag) {
  int gid = blockIdx.x * 256 + threadIdx.x;
  if (gid >= 5 * 16384) return;
  int widx = gid >> 14;
  int f    = gid & 16383;
  int e    = f & 15;
  int lane = (f >> 4) & 31;
  int kc   = (f >> 9) & 3;
  int nb   = (f >> 11) & 7;
  const float* w = (widx == 0) ? wq : (widx == 1) ? wk : (widx == 2) ? wv
                 : (widx == 3) ? wg : w2;
  int krow = kc * 32 + (e >> 3) * 16 + (lane >> 4) * 8 + (e & 7);
  int col  = nb * 16 + (lane & 15);
  frag[gid] = (_Float16)w[krow * DZC + col];
}

// ---------------------------------------------------------------------------
// K1: transpose z (C,L,L)->(token,C) + LayerNorm over C, write f16.
// Second pass re-reads z; whole tensor (33.5MB) is L2-resident (192MB L2).
// ---------------------------------------------------------------------------
__global__ void layernorm_kernel(const float* __restrict__ z,
                                 const float* __restrict__ ln_w,
                                 const float* __restrict__ ln_b,
                                 _Float16* __restrict__ zn) {
  int t = blockIdx.x * blockDim.x + threadIdx.x;  // 0..NT-1
  float s = 0.f, s2 = 0.f;
#pragma unroll 4
  for (int c = 0; c < DZC; ++c) {
    float v = z[(size_t)c * NT + t];
    s += v; s2 += v * v;
  }
  float mu   = s * (1.f / DZC);
  float var  = s2 * (1.f / DZC) - mu * mu;
  float rstd = rsqrtf(var + 1e-5f);
#pragma unroll 4
  for (int c = 0; c < DZC; ++c) {
    float v = z[(size_t)c * NT + t];
    zn[(size_t)t * DZC + c] = (_Float16)(((v - mu) * rstd) * ln_w[c] + ln_b[c]);
  }
}

// ---------------------------------------------------------------------------
// K2: Q/K/V/G projections. Block = 64 tokens (4 tiles of 16), 8 waves; wave nb
// owns output column block nb. All 16 B fragments (4 weights x 4 K-chunks,
// ~128 VGPRs) are held in registers and reused across the 4 token tiles,
// cutting weight L2 traffic 4x vs one-tile blocks.
// ---------------------------------------------------------------------------
__global__ __launch_bounds__(256) void qkvg_proj_kernel(
    const _Float16* __restrict__ zn, const _Float16* __restrict__ wfrag,
    const float* __restrict__ bq, const float* __restrict__ bk,
    const float* __restrict__ bv, const float* __restrict__ bg,
    _Float16* __restrict__ qh, _Float16* __restrict__ kh,
    _Float16* __restrict__ vh, _Float16* __restrict__ gh) {
  const int lane = threadIdx.x & 31;
  const int nb   = threadIdx.x >> 5;       // wave id = output column block
  const int half = lane >> 4, ml = lane & 15;

  v16h Bq[4], Bk[4], Bv[4], Bg[4];
#pragma unroll
  for (int kc = 0; kc < 4; ++kc) {
    int fidx = ((nb * 4 + kc) * 32 + lane) * 16;
    Bq[kc] = ld16(wfrag + 0 * 16384 + fidx);
    Bk[kc] = ld16(wfrag + 1 * 16384 + fidx);
    Bv[kc] = ld16(wfrag + 2 * 16384 + fidx);
    Bg[kc] = ld16(wfrag + 3 * 16384 + fidx);
  }

  const int col = nb * 16 + ml;
  const float bqv = bq[col], bkv = bk[col], bvv = bv[col], bgv = bg[col];
  const float qscale = 0.17677669529663687f;  // 1/sqrt(32)

#pragma unroll
  for (int tile = 0; tile < 4; ++tile) {
    const int t0 = blockIdx.x * 64 + tile * 16;
    v8f aq = {}, ak = {}, av = {}, ag = {};
#pragma unroll
    for (int kc = 0; kc < 4; ++kc) {
      v16h A = ld_a16(zn + (size_t)(t0 + ml) * DZC + kc * 32 + half * 8);
      aq = wmma_f16(A, Bq[kc], aq);
      ak = wmma_f16(A, Bk[kc], ak);
      av = wmma_f16(A, Bv[kc], av);
      ag = wmma_f16(A, Bg[kc], ag);
    }
#pragma unroll
    for (int r = 0; r < 8; ++r) {
      int token = t0 + r + half * 8;
      size_t o = (size_t)token * DZC + col;
      qh[o] = (_Float16)((aq[r] + bqv) * qscale);
      kh[o] = (_Float16)(ak[r] + bkv);
      vh[o] = (_Float16)(av[r] + bvv);
      float gx = ag[r] + bgv;
      gh[o] = (_Float16)(1.f / (1.f + __expf(-gx)));
    }
  }
}

// ---------------------------------------------------------------------------
// K3: attention per (n,h). K staged into LDS via CDNA5 async global->LDS copy
// (ASYNCcnt) when available; V^T via generic loads (needs transpose).
// Flash-style online softmax; wave handles 32 i-rows (2 tiles of 16); j in
// chunks of 32 so the P@V WMMA gets a full K=32.
// ---------------------------------------------------------------------------
__global__ __launch_bounds__(256) void attention_kernel(
    const _Float16* __restrict__ qh, const _Float16* __restrict__ kh,
    const _Float16* __restrict__ vh, const _Float16* __restrict__ gh,
    _Float16* __restrict__ zo) {
  __shared__ __align__(16) _Float16 k_sh[LSEQ * DCH];   // [j][c]   16 KB
  __shared__ __align__(16) _Float16 v_t[DCH * LSEQ];    // [c][j]   16 KB
  __shared__ __align__(16) _Float16 p_sh[8][16 * 32];   // per-wave  8 KB

  const int n = blockIdx.x >> 2;
  const int h = blockIdx.x & 3;
  const int tid = threadIdx.x;

#if USE_ASYNC_LDS
  // K tile: 256 rows x 64B -> 1024 b128 chunks, 4 per thread, async DMA to LDS.
  for (int idx = tid; idx < LSEQ * 4; idx += 256) {
    int j = idx >> 2, qo = idx & 3;
    const _Float16* g = kh + (size_t)(n * LSEQ + j) * DZC + h * DCH + qo * 8;
    __builtin_amdgcn_global_load_async_to_lds_b128(
        (__attribute__((address_space(1))) i4v*)(size_t)g,
        (__attribute__((address_space(3))) i4v*)&k_sh[j * 32 + qo * 8], 0, 0);
  }
  for (int idx = tid; idx < LSEQ * DCH; idx += 256) {
    int j = idx >> 5, c = idx & 31;
    v_t[c * LSEQ + j] = vh[(size_t)(n * LSEQ + j) * DZC + h * DCH + c];
  }
  asm volatile("s_wait_asynccnt 0x0" ::: "memory");
#else
  for (int idx = tid; idx < LSEQ * DCH; idx += 256) {
    int j = idx >> 5, c = idx & 31;
    size_t g = (size_t)(n * LSEQ + j) * DZC + h * DCH + c;
    k_sh[idx] = kh[g];
    v_t[c * LSEQ + j] = vh[g];
  }
#endif
  __syncthreads();

  const int lane = tid & 31, wid = tid >> 5;
  const int half = lane >> 4, ml = lane & 15;

  for (int ti = 0; ti < 2; ++ti) {
    const int i0 = (wid * 2 + ti) * 16;
    v16h qA = ld_a16(qh + (size_t)(n * LSEQ + i0 + ml) * DZC + h * DCH + half * 8);

    float m[8], lsum[8];
    v8f oc0 = {}, oc1 = {};
#pragma unroll
    for (int r = 0; r < 8; ++r) { m[r] = -1e30f; lsum[r] = 0.f; }

    for (int jc = 0; jc < 8; ++jc) {
      // scores: S(16i x 32j) = Q(16x32c) @ K^T; B frag K=c contiguous per row
      v16h kB0 = ld_a16(&k_sh[(jc * 32 + ml) * 32 + half * 8]);
      v16h kB1 = ld_a16(&k_sh[(jc * 32 + 16 + ml) * 32 + half * 8]);
      v8f zc = {};
      v8f s0 = wmma_f16(qA, kB0, zc);
      v8f s1 = wmma_f16(qA, kB1, zc);

#pragma unroll
      for (int r = 0; r < 8; ++r) {
        // row reductions stay inside 16-lane halves (masks 1,2,4,8)
        float rm = fmaxf(s0[r], s1[r]);
        rm = fmaxf(rm, __shfl_xor(rm, 1, 32));
        rm = fmaxf(rm, __shfl_xor(rm, 2, 32));
        rm = fmaxf(rm, __shfl_xor(rm, 4, 32));
        rm = fmaxf(rm, __shfl_xor(rm, 8, 32));
        float mn = fmaxf(m[r], rm);
        float sc = __expf(m[r] - mn);
        float p0 = __expf(s0[r] - mn);
        float p1 = __expf(s1[r] - mn);
        float rs = p0 + p1;
        rs += __shfl_xor(rs, 1, 32);
        rs += __shfl_xor(rs, 2, 32);
        rs += __shfl_xor(rs, 4, 32);
        rs += __shfl_xor(rs, 8, 32);
        lsum[r] = lsum[r] * sc + rs;
        m[r] = mn;
        oc0[r] *= sc;
        oc1[r] *= sc;
        // C-layout -> LDS (i-row major) for A-layout reload
        int row = r + half * 8;
        p_sh[wid][row * 32 + ml]      = (_Float16)p0;
        p_sh[wid][row * 32 + 16 + ml] = (_Float16)p1;
      }
      // P as A-matrix (16i x 32j), V^T rows as B fragments (K=j contiguous)
      v16h pA  = ld_a16(&p_sh[wid][ml * 32 + half * 8]);
      v16h vB0 = ld_a16(&v_t[ml * LSEQ + jc * 32 + half * 8]);
      v16h vB1 = ld_a16(&v_t[(16 + ml) * LSEQ + jc * 32 + half * 8]);
      oc0 = wmma_f16(pA, vB0, oc0);
      oc1 = wmma_f16(pA, vB1, oc1);
    }

#pragma unroll
    for (int r = 0; r < 8; ++r) {
      float inv = 1.f / lsum[r];
      int token = n * LSEQ + i0 + r + half * 8;
      size_t base = (size_t)token * DZC + h * DCH;
      float g0 = (float)gh[base + ml];
      float g1 = (float)gh[base + 16 + ml];
      zo[base + ml]      = (_Float16)(oc0[r] * inv * g0);
      zo[base + 16 + ml] = (_Float16)(oc1[r] * inv * g1);
    }
  }
}

// ---------------------------------------------------------------------------
// K4: output projection zo(NT x 128) @ w2 + b2, transposed store into (C,L,L).
// Same 64-token blocking as K2 to amortize weight fragment loads.
// ---------------------------------------------------------------------------
__global__ __launch_bounds__(256) void out_proj_kernel(
    const _Float16* __restrict__ zo, const _Float16* __restrict__ w2f,
    const float* __restrict__ b2, float* __restrict__ out) {
  const int lane = threadIdx.x & 31;
  const int nb   = threadIdx.x >> 5;
  const int half = lane >> 4, ml = lane & 15;

  v16h B2[4];
#pragma unroll
  for (int kc = 0; kc < 4; ++kc)
    B2[kc] = ld16(w2f + ((nb * 4 + kc) * 32 + lane) * 16);

  const int col = nb * 16 + ml;
  const float bb = b2[col];

#pragma unroll
  for (int tile = 0; tile < 4; ++tile) {
    const int t0 = blockIdx.x * 64 + tile * 16;
    v8f acc = {};
#pragma unroll
    for (int kc = 0; kc < 4; ++kc) {
      v16h A = ld_a16(zo + (size_t)(t0 + ml) * DZC + kc * 32 + half * 8);
      acc = wmma_f16(A, B2[kc], acc);
    }
#pragma unroll
    for (int r = 0; r < 8; ++r) {
      int token = t0 + r + half * 8;
      out[(size_t)col * NT + token] = acc[r] + bb;
    }
  }
}

// ---------------------------------------------------------------------------
extern "C" void kernel_launch(void* const* d_in, const int* in_sizes, int n_in,
                              void* d_out, int out_size, void* d_ws,
                              size_t ws_size, hipStream_t stream) {
  (void)in_sizes; (void)n_in; (void)out_size; (void)ws_size;

  const float* z    = (const float*)d_in[0];
  const float* ln_w = (const float*)d_in[1];
  const float* ln_b = (const float*)d_in[2];
  const float* wq   = (const float*)d_in[3];
  const float* bq   = (const float*)d_in[4];
  const float* wk   = (const float*)d_in[5];
  const float* bk   = (const float*)d_in[6];
  const float* wv   = (const float*)d_in[7];
  const float* bv   = (const float*)d_in[8];
  const float* wg   = (const float*)d_in[9];
  const float* bg   = (const float*)d_in[10];
  const float* w2   = (const float*)d_in[11];
  const float* b2   = (const float*)d_in[12];
  float* out        = (float*)d_out;

  // workspace layout (bytes)
  char* ws = (char*)d_ws;
  const size_t FRAG_OFF = 0;                       // 5 * 32768 = 160 KB
  const size_t ZN_OFF   = 5 * 32768;               // NT*128 f16 = 16 MB
  const size_t TOK_B    = (size_t)NT * DZC * sizeof(_Float16);
  const size_t Q_OFF  = ZN_OFF + TOK_B;
  const size_t K_OFF  = Q_OFF + TOK_B;
  const size_t V_OFF  = K_OFF + TOK_B;
  const size_t G_OFF  = V_OFF + TOK_B;
  const size_t ZO_OFF = G_OFF + TOK_B;

  _Float16* frag = (_Float16*)(ws + FRAG_OFF);
  _Float16* zn   = (_Float16*)(ws + ZN_OFF);
  _Float16* qhp  = (_Float16*)(ws + Q_OFF);
  _Float16* khp  = (_Float16*)(ws + K_OFF);
  _Float16* vhp  = (_Float16*)(ws + V_OFF);
  _Float16* ghp  = (_Float16*)(ws + G_OFF);
  _Float16* zop  = (_Float16*)(ws + ZO_OFF);

  convert_weights_kernel<<<dim3(320), dim3(256), 0, stream>>>(wq, wk, wv, wg,
                                                              w2, frag);
  layernorm_kernel<<<dim3(256), dim3(256), 0, stream>>>(z, ln_w, ln_b, zn);
  qkvg_proj_kernel<<<dim3(NT / 64), dim3(256), 0, stream>>>(
      zn, frag, bq, bk, bv, bg, qhp, khp, vhp, ghp);
  attention_kernel<<<dim3(LSEQ * NHEAD), dim3(256), 0, stream>>>(qhp, khp, vhp,
                                                                 ghp, zop);
  out_proj_kernel<<<dim3(NT / 64), dim3(256), 0, stream>>>(
      zop, frag + 4 * 16384, b2, out);
}
